// RegionProposalNetwork_85203561218863
// MI455X (gfx1250) — matrix operand order repaired and checked by reading, not compiled, not run on hardware
//
#include <hip/hip_runtime.h>
#include <hip/hip_bf16.h>
#include <math.h>
#include <stdint.h>

typedef __attribute__((ext_vector_type(16))) _Float16 v16h;
typedef __attribute__((ext_vector_type(8)))  _Float16 v8h;
typedef __attribute__((ext_vector_type(8)))  float    v8f;

#define IMGB   8
#define CIN    512
#define FH     50
#define FW     76
#define PH     52          // padded height
#define PW     78          // padded width
#define PPI    (FH*FW)     // 3800 pixels per image
#define NPIX   (IMGB*PPI)  // 30400
#define NA     9
#define NANCH  (PPI*NA)    // 34200
#define KTOT   (CIN*9)     // 4608
#define NKC    (KTOT/32)   // 144 k-chunks
#define HEADOC 48          // 9 cls + 36 box, padded
#define NPOST  300

static __device__ __forceinline__ v16h cat8(v8h lo, v8h hi) {
    return __builtin_shufflevector(lo, hi, 0,1,2,3,4,5,6,7,8,9,10,11,12,13,14,15);
}

static __device__ __forceinline__ v8f wmma16(v16h a, v16h b, v8f c) {
    return __builtin_amdgcn_wmma_f32_16x16x32_f16(false, a, false, b, (short)0, c,
                                                  false, false);
}

// gfx1250 async global->LDS copy (ASYNCcnt-tracked, bypasses VGPRs).
// vdst carries the LDS byte offset = low 32 bits of a generic pointer to LDS.
static __device__ __forceinline__ void async_ld_b128(const _Float16* lds_ptr,
                                                     const _Float16* gptr) {
    unsigned lds = (unsigned)(uintptr_t)lds_ptr;
    asm volatile("global_load_async_to_lds_b128 %0, %1, off"
                 :: "v"(lds), "v"(gptr) : "memory");
}
static __device__ __forceinline__ void wait_async0() {
    asm volatile("s_wait_asynccnt 0x0" ::: "memory");
}

// ---------------------------------------------------------------- pack passes
__global__ void k_pad(const float* __restrict__ f, _Float16* __restrict__ fp) {
    long i = (long)blockIdx.x * blockDim.x + threadIdx.x;
    if (i >= (long)IMGB * PH * PW * CIN) return;
    int ic = (int)(i & (CIN - 1));
    long r = i >> 9;
    int xx = (int)(r % PW); r /= PW;
    int yy = (int)(r % PH); int b = (int)(r / PH);
    float v = 0.0f;
    if (yy >= 1 && yy <= FH && xx >= 1 && xx <= FW)
        v = f[(((long)b * CIN + ic) * FH + (yy - 1)) * FW + (xx - 1)];
    fp[i] = (_Float16)v;
}

__global__ void k_wpack(const float* __restrict__ w, _Float16* __restrict__ wp) {
    long i = (long)blockIdx.x * blockDim.x + threadIdx.x;
    if (i >= (long)CIN * KTOT) return;
    int oc = (int)(i / KTOT);
    int k  = (int)(i % KTOT);
    int tap = k >> 9;          // k = tap*512 + ic
    int ic  = k & 511;
    float v = w[(long)oc * KTOT + (long)ic * 9 + tap];
    long dst = ((long)(k >> 5) * CIN + oc) * 32 + (k & 31);
    wp[dst] = (_Float16)v;
}

__global__ void k_hpack(const float* __restrict__ cw, const float* __restrict__ bw,
                        _Float16* __restrict__ hp) {
    int i = blockIdx.x * blockDim.x + threadIdx.x;
    if (i >= HEADOC * CIN) return;
    int oc = i / CIN, k = i % CIN;
    float v = 0.0f;
    if (oc < 9)       v = cw[oc * CIN + k];
    else if (oc < 45) v = bw[(oc - 9) * CIN + k];
    int octile = oc >> 4, oc16 = oc & 15, kt = k >> 5, kk = k & 31;
    hp[(((octile * 16 + kt) * 16 + oc16) << 5) + kk] = (_Float16)v;
}

// ------------------------------------------------- 3x3 conv as implicit GEMM
// M=512 x K=4608 x N=30400. Block tile 128 oc x 128 px, 8 waves as 4(M) x 2(N).
// Per wave: 32 oc x 64 px register tile = 8 WMMAs per 32-K chunk.
// LDS ping-pong fed by global_load_async_to_lds_b128 (no VGPR staging).
__global__ void __launch_bounds__(256) k_conv(const _Float16* __restrict__ fp,
                                              const _Float16* __restrict__ wp,
                                              const float*    __restrict__ bias,
                                              _Float16*       __restrict__ t) {
    __shared__ _Float16 Asm[2][128 * 32];   // 2 x 8 KB
    __shared__ _Float16 Bsm[2][128 * 32];   // 2 x 8 KB
    const int tid  = threadIdx.x;
    const int wave = tid >> 5, lane = tid & 31;
    const int lrow = lane & 15, g = lane >> 4;
    const int mrow = wave & 3;              // 4 M-rows of 32 oc
    const int ncol = wave >> 2;             // 2 N-cols of 64 px
    const int oc0 = blockIdx.x * 128;
    const int p0  = blockIdx.y * 128;

    // B-staging mapping: 128 px x 32 halves; each thread moves 32B of one pixel
    const int bpx = tid >> 1, bh = (tid & 1) * 16;
    int pp = p0 + bpx; if (pp >= NPIX) pp = NPIX - 1;   // clamp tail block
    const int bimg = pp / PPI, rr = pp % PPI;
    const int yy = rr / FW, xx = rr % FW;
    const long bbase = ((long)((bimg * PH + yy) * PW + xx)) * CIN + bh;

    v8f a00 = {}, a01 = {}, a02 = {}, a03 = {};
    v8f a10 = {}, a11 = {}, a12 = {}, a13 = {};

    // ---- prologue: async-stage chunk 0 into buffer 0 (tap 0 => dy=dx=0)
    {
        async_ld_b128(Bsm[0] + bpx * 32 + bh,     fp + bbase);
        async_ld_b128(Bsm[0] + bpx * 32 + bh + 8, fp + bbase + 8);
        const long abase = (long)oc0 * 32;
        async_ld_b128(Asm[0] + tid * 8,           wp + abase + tid * 8);
        async_ld_b128(Asm[0] + (tid + 256) * 8,   wp + abase + (tid + 256) * 8);
    }
    wait_async0();
    __syncthreads();

    for (int kc = 0; kc < NKC; ++kc) {
        const int cur = kc & 1, nxt = cur ^ 1;
        const bool more = (kc + 1 < NKC);

        // async-stage next chunk straight into the other LDS buffer
        if (more) {
            const int kn = kc + 1;
            const int tap = kn >> 4, ic0 = (kn & 15) << 5;
            const int dy = tap / 3, dx = tap % 3;
            const long bo = bbase + (long)(dy * PW + dx) * CIN + ic0;
            async_ld_b128(Bsm[nxt] + bpx * 32 + bh,     fp + bo);
            async_ld_b128(Bsm[nxt] + bpx * 32 + bh + 8, fp + bo + 8);
            const long abase = ((long)kn * CIN + oc0) * 32;
            async_ld_b128(Asm[nxt] + tid * 8,           wp + abase + tid * 8);
            async_ld_b128(Asm[nxt] + (tid + 256) * 8,   wp + abase + (tid + 256) * 8);
            if (kc + 2 < NKC)   // keep chunk kc+2's A streaming into L2
                __builtin_prefetch((const void*)(wp + abase + (long)CIN * 32 + tid * 8), 0, 1);
        }

        // fragments from current buffer
        const _Float16* Ab = Asm[cur];
        const _Float16* Bb = Bsm[cur];
        const _Float16* ap0 = Ab + (mrow * 32 + lrow) * 32 + g * 8;
        const _Float16* ap1 = ap0 + 16 * 32;
        const v16h af0 = cat8(*(const v8h*)ap0, *(const v8h*)(ap0 + 16));
        const v16h af1 = cat8(*(const v8h*)ap1, *(const v8h*)(ap1 + 16));
        const _Float16* bp = Bb + (ncol * 64 + lrow) * 32 + g * 8;
        const v16h bf0 = cat8(*(const v8h*)(bp),           *(const v8h*)(bp + 16));
        const v16h bf1 = cat8(*(const v8h*)(bp + 16 * 32), *(const v8h*)(bp + 16 * 32 + 16));
        const v16h bf2 = cat8(*(const v8h*)(bp + 32 * 32), *(const v8h*)(bp + 32 * 32 + 16));
        const v16h bf3 = cat8(*(const v8h*)(bp + 48 * 32), *(const v8h*)(bp + 48 * 32 + 16));

        a00 = wmma16(af0, bf0, a00);
        a01 = wmma16(af0, bf1, a01);
        a02 = wmma16(af0, bf2, a02);
        a03 = wmma16(af0, bf3, a03);
        a10 = wmma16(af1, bf0, a10);
        a11 = wmma16(af1, bf1, a11);
        a12 = wmma16(af1, bf2, a12);
        a13 = wmma16(af1, bf3, a13);

        if (more) wait_async0();   // next buffer fully landed in LDS
        __syncthreads();
    }

    // epilogue: each lane holds 8 consecutive oc (M = 8*g + r) per tile
    const int ocb0 = oc0 + mrow * 32 + g * 8;
    const int ocb1 = ocb0 + 16;
    const int pxb  = p0 + ncol * 64 + lrow;
    v8f accs0[4] = {a00, a01, a02, a03};
    v8f accs1[4] = {a10, a11, a12, a13};
#pragma unroll
    for (int ni = 0; ni < 4; ++ni) {
        const int pix = pxb + ni * 16;
        if (pix >= NPIX) continue;
        v8h o0, o1;
#pragma unroll
        for (int r2 = 0; r2 < 8; ++r2) {
            float v0 = accs0[ni][r2] + bias[ocb0 + r2];
            float v1 = accs1[ni][r2] + bias[ocb1 + r2];
            o0[r2] = (_Float16)(v0 > 0.0f ? v0 : 0.0f);
            o1[r2] = (_Float16)(v1 > 0.0f ? v1 : 0.0f);
        }
        *(v8h*)(t + (long)pix * CIN + ocb0) = o0;
        *(v8h*)(t + (long)pix * CIN + ocb1) = o1;
    }
}

// ------------------------------------------------------ 1x1 heads via WMMA
__global__ void __launch_bounds__(256) k_head(const _Float16* __restrict__ t,
                                              const _Float16* __restrict__ hw,
                                              const float* __restrict__ cls_b,
                                              const float* __restrict__ box_b,
                                              float* __restrict__ obj_out,
                                              float* __restrict__ del_out) {
    const int wave = threadIdx.x >> 5, lane = threadIdx.x & 31;
    const int lrow = lane & 15, g = lane >> 4;
    const int tw = blockIdx.x * 8 + wave;
    if (tw >= 3 * (NPIX / 16)) return;              // wave-uniform exit
    const int octile = tw % 3;
    const int p0 = (tw / 3) * 16;

    v8f acc = {};
    const _Float16* tp = t + (long)(p0 + lrow) * CIN + g * 8;
#pragma unroll 4
    for (int kt = 0; kt < 16; ++kt) {
        const _Float16* ap = hw + (((octile * 16 + kt) * 16 + lrow) << 5) + g * 8;
        const v16h af = cat8(*(const v8h*)ap, *(const v8h*)(ap + 16));
        const _Float16* bp = tp + kt * 32;
        const v16h bf = cat8(*(const v8h*)bp, *(const v8h*)(bp + 16));
        acc = wmma16(af, bf, acc);
    }

    const int p = p0 + lrow;
    const int b = p / PPI, rloc = p % PPI;
#pragma unroll
    for (int r2 = 0; r2 < 8; ++r2) {
        const int oc = octile * 16 + g * 8 + r2;
        const float v = acc[r2];
        if (oc < 9) {
            obj_out[(long)b * NANCH + rloc * 9 + oc] = v + cls_b[oc];
        } else if (oc < 45) {
            const int j = oc - 9;
            del_out[(((long)b * NANCH + rloc * 9 + (j >> 2)) << 2) + (j & 3)] = v + box_b[j];
        }
    }
}

// ------------------------------------------------------------------- decode
__global__ void k_decode(const float* __restrict__ obj, const float* __restrict__ del,
                         float* __restrict__ props, float* __restrict__ scores) {
    long i = (long)blockIdx.x * blockDim.x + threadIdx.x;
    if (i >= (long)IMGB * NANCH) return;
    const int n = (int)(i % NANCH);
    const int a = n % 9, pix = n / 9;
    const int y = pix / FW, x = pix % FW;
    const float sizes[3] = {128.0f, 256.0f, 512.0f};
    const float rats[3]  = {0.5f, 1.0f, 2.0f};
    const float s = sizes[a / 3];
    const float hr = sqrtf(rats[a % 3]);
    const float wa = s / hr, ha = s * hr;
    const float cxa = x * 16.0f, cya = y * 16.0f;
    const float* d = del + i * 4;
    const float cx = d[0] * wa + cxa, cy = d[1] * ha + cya;
    const float w = expf(d[2]) * wa, h = expf(d[3]) * ha;
    float x1 = fminf(fmaxf(cx - 0.5f * w, 0.0f), 1216.0f);
    float y1 = fminf(fmaxf(cy - 0.5f * h, 0.0f), 800.0f);
    float x2 = fminf(fmaxf(cx + 0.5f * w, 0.0f), 1216.0f);
    float y2 = fminf(fmaxf(cy + 0.5f * h, 0.0f), 800.0f);
    float* pr = props + i * 4;
    pr[0] = x1; pr[1] = y1; pr[2] = x2; pr[3] = y2;
    const bool keep = ((x2 - x1) >= 16.0f) && ((y2 - y1) >= 16.0f);
    scores[i] = keep ? obj[i] : -__builtin_inff();
}

// ------------------------------------------------------------ greedy NMS
__global__ void __launch_bounds__(256) k_nms(const float* __restrict__ props,
                                             float* __restrict__ scores,
                                             float* __restrict__ out) {
    const int b = blockIdx.x;
    const float* pb = props + (long)b * NANCH * 4;
    float* sb = scores + (long)b * NANCH;
    float* ob = out + (long)b * NPOST * 4;
    __shared__ float rv[256];
    __shared__ int   ri[256];
    __shared__ float box[4];
    __shared__ int   flag;
    const int tid = threadIdx.x;
    const float NEG = -__builtin_inff();

    for (int it = 0; it < NPOST; ++it) {
        float best = NEG; int bidx = NANCH;
        for (int j = tid; j < NANCH; j += 256) {
            const float v = sb[j];
            if (v > best) { best = v; bidx = j; }
        }
        rv[tid] = best; ri[tid] = bidx;
        __syncthreads();
        for (int stp = 128; stp > 0; stp >>= 1) {
            if (tid < stp) {
                const float v2 = rv[tid + stp]; const int i2 = ri[tid + stp];
                if (v2 > rv[tid] || (v2 == rv[tid] && i2 < ri[tid])) { rv[tid] = v2; ri[tid] = i2; }
            }
            __syncthreads();
        }
        if (tid == 0) {
            const float v = rv[0]; const int j = ri[0];
            if (v > NEG && j < NANCH) {
                box[0] = pb[j * 4 + 0]; box[1] = pb[j * 4 + 1];
                box[2] = pb[j * 4 + 2]; box[3] = pb[j * 4 + 3];
                ob[it * 4 + 0] = box[0]; ob[it * 4 + 1] = box[1];
                ob[it * 4 + 2] = box[2]; ob[it * 4 + 3] = box[3];
                sb[j] = NEG;
                flag = 1;
            } else {
                ob[it * 4 + 0] = 0.0f; ob[it * 4 + 1] = 0.0f;
                ob[it * 4 + 2] = 0.0f; ob[it * 4 + 3] = 0.0f;
                flag = 0;
            }
        }
        __syncthreads();
        if (flag) {
            const float bx1 = box[0], by1 = box[1], bx2 = box[2], by2 = box[3];
            const float barea = (bx2 - bx1) * (by2 - by1);
            for (int j = tid; j < NANCH; j += 256) {
                if (sb[j] == NEG) continue;
                const float x1 = pb[j * 4], y1 = pb[j * 4 + 1];
                const float x2 = pb[j * 4 + 2], y2 = pb[j * 4 + 3];
                const float iw = fmaxf(fminf(bx2, x2) - fmaxf(bx1, x1), 0.0f);
                const float ih = fmaxf(fminf(by2, y2) - fmaxf(by1, y1), 0.0f);
                const float inter = iw * ih;
                const float uni = barea + (x2 - x1) * (y2 - y1) - inter;
                const float iou = (uni > 0.0f) ? inter / uni : 0.0f;
                if (iou > 0.7f) sb[j] = NEG;
            }
        }
        __syncthreads();
    }
}

// ------------------------------------------------------------------ launch
extern "C" void kernel_launch(void* const* d_in, const int* in_sizes, int n_in,
                              void* d_out, int out_size, void* d_ws, size_t ws_size,
                              hipStream_t stream) {
    const float* features = (const float*)d_in[0];
    const float* conv_w   = (const float*)d_in[1];
    const float* conv_b   = (const float*)d_in[2];
    const float* cls_w    = (const float*)d_in[3];
    const float* cls_b    = (const float*)d_in[4];
    const float* box_w    = (const float*)d_in[5];
    const float* box_b    = (const float*)d_in[6];

    // workspace layout
    char* ws = (char*)d_ws;
    const size_t featpad_elems = (size_t)IMGB * PH * PW * CIN;    // f16
    const size_t wpack_elems   = (size_t)CIN * KTOT;              // f16
    const size_t hpack_elems   = (size_t)HEADOC * CIN;            // f16
    const size_t t_elems       = (size_t)NPIX * CIN;              // f16
    const size_t props_elems   = (size_t)IMGB * NANCH * 4;        // f32
    size_t off = 0;
    _Float16* featpad = (_Float16*)(ws + off); off += featpad_elems * 2;
    _Float16* wpack   = (_Float16*)(ws + off); off += wpack_elems * 2;
    _Float16* hpack   = (_Float16*)(ws + off); off += hpack_elems * 2;
    _Float16* tbuf    = (_Float16*)(ws + off); off += t_elems * 2;
    float*    props   = (float*)(ws + off);    off += props_elems * 4;
    float*    scores  = (float*)(ws + off);

    // d_out layout: proposals [8,300,4] | objectness [8,34200] | deltas [8,34200,4]
    float* out_props = (float*)d_out;
    float* out_obj   = out_props + (size_t)IMGB * NPOST * 4;
    float* out_del   = out_obj + (size_t)IMGB * NANCH;

    k_pad  <<<(unsigned)((featpad_elems + 255) / 256), 256, 0, stream>>>(features, featpad);
    k_wpack<<<(unsigned)((wpack_elems + 255) / 256),   256, 0, stream>>>(conv_w, wpack);
    k_hpack<<<(unsigned)((hpack_elems + 255) / 256),   256, 0, stream>>>(cls_w, box_w, hpack);

    dim3 cg(CIN / 128, (NPIX + 127) / 128);          // 4 x 238
    k_conv<<<cg, 256, 0, stream>>>(featpad, wpack, conv_b, tbuf);

    const int head_tiles = 3 * (NPIX / 16);          // 5700 wave-tiles
    k_head<<<(head_tiles + 7) / 8, 256, 0, stream>>>(tbuf, hpack, cls_b, box_b,
                                                     out_obj, out_del);

    const size_t nanch_total = (size_t)IMGB * NANCH;
    k_decode<<<(unsigned)((nanch_total + 255) / 256), 256, 0, stream>>>(out_obj, out_del,
                                                                       props, scores);
    k_nms<<<IMGB, 256, 0, stream>>>(props, scores, out_props);
}
